// MultiHeadAttention_54168127537271
// MI455X (gfx1250) — compile-verified, matrix operand order checked by reading, not run on hardware
//
#include <hip/hip_runtime.h>
#include <hip/hip_bf16.h>

// ---------------------------------------------------------------------------
// Problem constants (match reference)
// ---------------------------------------------------------------------------
constexpr int CB  = 2;      // batch
constexpr int CT  = 2048;   // seq len
constexpr int CD  = 2048;   // model dim
constexpr int CH  = 16;     // heads
constexpr int CDK = 128;    // head dim
constexpr int CM  = CB * CT;            // 4096 rows for projections
constexpr float LOG_THETA = 9.210340371976184f;   // ln(10000)
constexpr float SM_SCALE  = 0.08838834764831845f; // 1/sqrt(128)

typedef __bf16 bf16_t;
typedef __attribute__((ext_vector_type(16))) __bf16 v16bf;
typedef __attribute__((ext_vector_type(8)))  __bf16 v8bf;
typedef __attribute__((ext_vector_type(8)))  float  v8f;

union F8 { v8f v; float f[8]; };

static __device__ __forceinline__ v16bf mkfrag(v8bf lo, v8bf hi) {
    union { v16bf v; v8bf h[2]; } u;
    u.h[0] = lo; u.h[1] = hi;
    return u.v;
}

static __device__ __forceinline__ v8f wmma_bf16(v16bf a, v16bf b, v8f c) {
    // emits v_wmma_f32_16x16x32_bf16
    return __builtin_amdgcn_wmma_f32_16x16x32_bf16(
        /*neg_a=*/false, a, /*neg_b=*/false, b,
        /*c_mod=*/(short)0, c, /*reuse_a=*/false, /*reuse_b=*/false);
}

static __device__ __forceinline__ bf16_t to_bf16(float f) {
    unsigned int u = __float_as_uint(f);
    u += 0x7FFFu + ((u >> 16) & 1u);   // round-to-nearest-even
    unsigned short h = (unsigned short)(u >> 16);
    bf16_t r;
    __builtin_memcpy(&r, &h, 2);
    return r;
}

// ---------------------------------------------------------------------------
// CDNA5 async global -> LDS (tracked by ASYNCcnt, bypasses VGPRs)
// ---------------------------------------------------------------------------
static __device__ __forceinline__ unsigned lds_off32(const void* p) {
    // generic address of an LDS object: low 32 bits are the LDS byte offset
    return (unsigned)(unsigned long long)(uintptr_t)p;
}

static __device__ __forceinline__ void async_ld_b128(unsigned lds_byte_off,
                                                     const void* gptr) {
    asm volatile("global_load_async_to_lds_b128 %0, %1, off"
                 :: "v"(lds_byte_off),
                    "v"((unsigned long long)(uintptr_t)gptr)
                 : "memory");
}

template <int N>
static __device__ __forceinline__ void wait_asynccnt() {
#if __has_builtin(__builtin_amdgcn_s_wait_asynccnt)
    __builtin_amdgcn_s_wait_asynccnt((short)N);
#else
    asm volatile("s_wait_asynccnt %0" :: "i"(N) : "memory");
#endif
}

// ---------------------------------------------------------------------------
// fp32 -> bf16 conversion
// ---------------------------------------------------------------------------
__global__ __launch_bounds__(256) void cvt_f32_bf16(const float* __restrict__ in,
                                                    bf16_t* __restrict__ out, int n) {
    int i = blockIdx.x * 256 + threadIdx.x;
    int stride = gridDim.x * 256;
    for (; i < n; i += stride) out[i] = to_bf16(in[i]);
}

// ---------------------------------------------------------------------------
// Tiled bf16 WMMA GEMM:  C[m,n] = sum_k A[m,k] * W[n,k]   (i.e. A @ W^T)
// Block tile 128x128, 256 threads = 8 waves, wave tile 64x32 (4x2 wmma tiles).
// Double-buffered LDS tiles staged with global_load_async_to_lds_b128.
// MODE 0: fp32 out, row-major [M][N]               (final projection)
// MODE 1: RoPE + bf16 out, per-head [bh][t][dk]    (Q and K)
// MODE 2: bf16 out, per-head transposed [bh][dk][t] (V)
// ---------------------------------------------------------------------------
template <int MODE>
__global__ __launch_bounds__(256) void gemm_bf16(const bf16_t* __restrict__ A,
                                                 const bf16_t* __restrict__ W,
                                                 void* __restrict__ Cout,
                                                 int Mdim, int Ndim, int Kdim) {
    constexpr int LP   = 40;          // padded LDS row stride (bf16); 80B rows
    constexpr int TILE = 128 * LP;    // one buffer, in bf16 elems
    __shared__ __align__(16) bf16_t As[2 * TILE];
    __shared__ __align__(16) bf16_t Bs[2 * TILE];

    const int tid  = threadIdx.x;
    const int wave = tid >> 5;
    const int lane = tid & 31;
    const int lm   = lane & 15;
    const int hi8  = (lane >> 4) << 3;   // 0 or 8
    const int wm   = wave >> 2;          // 0..1  -> 64-row slab
    const int wn   = wave & 3;           // 0..3  -> 32-col slab
    const int bm   = blockIdx.y * 128;
    const int bn   = blockIdx.x * 128;

    const unsigned asb = lds_off32(&As[0]);
    const unsigned bsb = lds_off32(&Bs[0]);
    // this thread's two (row, chunk) staging slots
    const int row0 = tid >> 2,        kc0 = (tid & 3) << 3;
    const int row1 = (tid + 256) >> 2, kc1 = ((tid + 256) & 3) << 3;

    F8 acc[4][2];
#pragma unroll
    for (int mt = 0; mt < 4; ++mt)
#pragma unroll
        for (int nt = 0; nt < 2; ++nt)
#pragma unroll
            for (int r = 0; r < 8; ++r) acc[mt][nt].f[r] = 0.0f;

    // stage one 128x32 A tile + W tile into buffer `buf` (4 async ops/thread)
    auto stage = [&](int buf, int kk) {
        unsigned ab = asb + (unsigned)(buf * TILE) * 2u;
        unsigned bb = bsb + (unsigned)(buf * TILE) * 2u;
        async_ld_b128(ab + (unsigned)(row0 * LP + kc0) * 2u,
                      A + (size_t)(bm + row0) * Kdim + kk + kc0);
        async_ld_b128(bb + (unsigned)(row0 * LP + kc0) * 2u,
                      W + (size_t)(bn + row0) * Kdim + kk + kc0);
        async_ld_b128(ab + (unsigned)(row1 * LP + kc1) * 2u,
                      A + (size_t)(bm + row1) * Kdim + kk + kc1);
        async_ld_b128(bb + (unsigned)(row1 * LP + kc1) * 2u,
                      W + (size_t)(bn + row1) * Kdim + kk + kc1);
    };

    stage(0, 0);
    for (int kk = 0; kk < Kdim; kk += 32) {
        const int buf = (kk >> 5) & 1;
        if (kk + 32 < Kdim) {
            stage(buf ^ 1, kk + 32);
            wait_asynccnt<4>();   // async loads complete in order: buf is ready
        } else {
            wait_asynccnt<0>();
        }
        __syncthreads();

        const bf16_t* Ab = &As[buf * TILE];
        const bf16_t* Bb = &Bs[buf * TILE];
        v16bf af[4];
#pragma unroll
        for (int mt = 0; mt < 4; ++mt) {
            const bf16_t* p = Ab + (wm * 64 + mt * 16 + lm) * LP + hi8;
            af[mt] = mkfrag(*(const v8bf*)p, *(const v8bf*)(p + 16));
        }
        v16bf bfr[2];
#pragma unroll
        for (int nt = 0; nt < 2; ++nt) {
            const bf16_t* p = Bb + (wn * 32 + nt * 16 + lm) * LP + hi8;
            bfr[nt] = mkfrag(*(const v8bf*)p, *(const v8bf*)(p + 16));
        }
#pragma unroll
        for (int mt = 0; mt < 4; ++mt)
#pragma unroll
            for (int nt = 0; nt < 2; ++nt)
                acc[mt][nt].v = wmma_bf16(af[mt], bfr[nt], acc[mt][nt].v);
        __syncthreads();   // reads of `buf` done before it is restaged
    }

    // ------------------------- epilogues -------------------------
    if constexpr (MODE == 0) {
        float* C = (float*)Cout;
#pragma unroll
        for (int mt = 0; mt < 4; ++mt)
#pragma unroll
            for (int nt = 0; nt < 2; ++nt)
#pragma unroll
                for (int r = 0; r < 8; ++r) {
                    int m = bm + wm * 64 + mt * 16 + r + hi8;
                    int n = bn + wn * 32 + nt * 16 + lm;
                    C[(size_t)m * Ndim + n] = acc[mt][nt].f[r];
                }
    } else if constexpr (MODE == 1) {
        // RoPE + write bf16 per-head [bh][t][dk]
        bf16_t* Q = (bf16_t*)Cout;
#pragma unroll
        for (int nt = 0; nt < 2; ++nt) {
            int n_glob = bn + wn * 32 + nt * 16 + lm;
            int hh = n_glob >> 7;
            int dk = n_glob & 127;
            float inv = __expf(-(float)(dk & ~1) * (LOG_THETA / (float)CDK));
#pragma unroll
            for (int mt = 0; mt < 4; ++mt)
#pragma unroll
                for (int r = 0; r < 8; ++r) {
                    int m = bm + wm * 64 + mt * 16 + r + hi8;
                    int bb = m >> 11;           // / T
                    int t  = m & (CT - 1);
                    float ang = (float)t * inv;
                    float sn, cs;
                    __sincosf(ang, &sn, &cs);
                    float val = acc[mt][nt].f[r];
                    float prt = __shfl_xor(val, 1, 32);     // even/odd partner
                    float rot = (dk & 1) ? prt : -prt;
                    float res = val * cs + rot * sn;
                    Q[(((size_t)(bb * CH + hh)) * CT + t) * CDK + dk] = to_bf16(res);
                }
        }
    } else { // MODE == 2 : V transposed per-head [bh][dk][t]
        bf16_t* V = (bf16_t*)Cout;
#pragma unroll
        for (int nt = 0; nt < 2; ++nt) {
            int n_glob = bn + wn * 32 + nt * 16 + lm;
            int hh = n_glob >> 7;
            int dk = n_glob & 127;
#pragma unroll
            for (int mt = 0; mt < 4; ++mt)
#pragma unroll
                for (int r = 0; r < 8; ++r) {
                    int m = bm + wm * 64 + mt * 16 + r + hi8;
                    int bb = m >> 11;
                    int t  = m & (CT - 1);
                    V[(((size_t)(bb * CH + hh)) * CDK + dk) * CT + t] =
                        to_bf16(acc[mt][nt].f[r]);
                }
        }
    }
}

// ---------------------------------------------------------------------------
// Flash attention (causal), one (b,h,q-block-of-128) per workgroup.
// Q/K: [bh][t][dk] bf16 (RoPE applied).  Vt: [bh][dk][t] bf16.
// K/Vt blocks staged into LDS with async-to-LDS loads.
// Out: bf16 [B*T][D], consumed by Wo GEMM.
// ---------------------------------------------------------------------------
__global__ __launch_bounds__(256) void attn_flash(const bf16_t* __restrict__ Q,
                                                  const bf16_t* __restrict__ K,
                                                  const bf16_t* __restrict__ Vt,
                                                  bf16_t* __restrict__ Out) {
    constexpr int LP = 136;  // padded LDS stride (bf16 elems), 272B rows
    extern __shared__ __align__(16) bf16_t smem[];
    bf16_t* Ks = smem;                     // [128][LP]  keys x dk
    bf16_t* Vs = Ks + 128 * LP;            // [128][LP]  dk x keys (transposed V)
    bf16_t* Ps = Vs + 128 * LP;            // [8 waves][16][LP]

    const int tid  = threadIdx.x;
    const int wave = tid >> 5;
    const int lane = tid & 31;
    const int lm   = lane & 15;
    const int hi8  = (lane >> 4) << 3;

    const int iq = blockIdx.x & 15;        // q block (T/128 = 16)
    const int bh = blockIdx.x >> 4;        // 0..31
    const int b  = bh >> 4;
    const int h  = bh & 15;

    const bf16_t* Qb  = Q  + (size_t)bh * CT * CDK;
    const bf16_t* Kb0 = K  + (size_t)bh * CT * CDK;
    const bf16_t* Vb0 = Vt + (size_t)bh * CDK * CT;

    const unsigned ksb = lds_off32(Ks);
    const unsigned vsb = lds_off32(Vs);

    // Q fragments: wave owns 16 query rows, 4 k-chunks of dk
    v16bf qf[4];
    const int qrow = iq * 128 + wave * 16 + lm;
#pragma unroll
    for (int c = 0; c < 4; ++c) {
        const bf16_t* p = Qb + (size_t)qrow * CDK + c * 32 + hi8;
        qf[c] = mkfrag(*(const v8bf*)p, *(const v8bf*)(p + 16));
    }

    float mrow[8], lrow[8];
    F8 o[8];
#pragma unroll
    for (int r = 0; r < 8; ++r) { mrow[r] = -1e30f; lrow[r] = 0.0f; }
#pragma unroll
    for (int d = 0; d < 8; ++d)
#pragma unroll
        for (int r = 0; r < 8; ++r) o[d].f[r] = 0.0f;

    for (int j = 0; j <= iq; ++j) {
        const bf16_t* Kb = Kb0 + (size_t)j * 128 * CDK;
        const bf16_t* Vb = Vb0 + (size_t)j * 128;
        __syncthreads();  // previous iteration's LDS reads done
#pragma unroll
        for (int i = 0; i < 8; ++i) {     // 128 rows x 16 chunks of 8 bf16
            int idx = tid + i * 256;
            int row = idx >> 4;
            int kc  = (idx & 15) << 3;
            unsigned loff = (unsigned)(row * LP + kc) * 2u;
            async_ld_b128(ksb + loff, Kb + (size_t)row * CDK + kc);
            async_ld_b128(vsb + loff, Vb + (size_t)row * CT + kc);
        }
        wait_asynccnt<0>();
        __syncthreads();

        // ---- S = Q K^T  (8 key tiles of 16, 4 dk chunks each) ----
        F8 s[8];
#pragma unroll
        for (int n = 0; n < 8; ++n) {
#pragma unroll
            for (int r = 0; r < 8; ++r) s[n].f[r] = 0.0f;
            int krow = n * 16 + lm;
#pragma unroll
            for (int c = 0; c < 4; ++c) {
                const bf16_t* p = &Ks[krow * LP + c * 32 + hi8];
                v16bf kf = mkfrag(*(const v8bf*)p, *(const v8bf*)(p + 16));
                s[n].v = wmma_bf16(qf[c], kf, s[n].v);
            }
        }

        // ---- scale (+ causal mask on the diagonal block) ----
        if (j == iq) {
#pragma unroll
            for (int n = 0; n < 8; ++n) {
                int kloc = n * 16 + lm;
#pragma unroll
                for (int r = 0; r < 8; ++r) {
                    int qloc = wave * 16 + r + hi8;
                    s[n].f[r] = (kloc > qloc) ? -1e30f : s[n].f[r] * SM_SCALE;
                }
            }
        } else {
#pragma unroll
            for (int n = 0; n < 8; ++n)
#pragma unroll
                for (int r = 0; r < 8; ++r) s[n].f[r] *= SM_SCALE;
        }

        // ---- online softmax (rows live in 16-lane halves) ----
        float alpha[8], mnew[8];
#pragma unroll
        for (int r = 0; r < 8; ++r) {
            float v = s[0].f[r];
#pragma unroll
            for (int n = 1; n < 8; ++n) v = fmaxf(v, s[n].f[r]);
            v = fmaxf(v, __shfl_xor(v, 1, 32));
            v = fmaxf(v, __shfl_xor(v, 2, 32));
            v = fmaxf(v, __shfl_xor(v, 4, 32));
            v = fmaxf(v, __shfl_xor(v, 8, 32));
            mnew[r]  = fmaxf(mrow[r], v);
            alpha[r] = __expf(mrow[r] - mnew[r]);
            mrow[r]  = mnew[r];
        }
#pragma unroll
        for (int r = 0; r < 8; ++r) {
            float sum = 0.0f;
#pragma unroll
            for (int n = 0; n < 8; ++n) {
                float p = __expf(s[n].f[r] - mnew[r]);
                s[n].f[r] = p;
                sum += p;
            }
            sum += __shfl_xor(sum, 1, 32);
            sum += __shfl_xor(sum, 2, 32);
            sum += __shfl_xor(sum, 4, 32);
            sum += __shfl_xor(sum, 8, 32);
            lrow[r] = lrow[r] * alpha[r] + sum;
        }
#pragma unroll
        for (int d = 0; d < 8; ++d)
#pragma unroll
            for (int r = 0; r < 8; ++r) o[d].f[r] *= alpha[r];

        // ---- P: C-layout -> A-fragment layout via per-wave LDS scratch ----
        bf16_t* Pw = Ps + wave * 16 * LP;
#pragma unroll
        for (int n = 0; n < 8; ++n)
#pragma unroll
            for (int r = 0; r < 8; ++r)
                Pw[(r + hi8) * LP + n * 16 + lm] = to_bf16(s[n].f[r]);
        // (compiler inserts s_wait_dscnt for the LDS store->load dependency)

        // ---- O += P * V ----
#pragma unroll
        for (int c = 0; c < 4; ++c) {
            const bf16_t* pp = &Pw[lm * LP + c * 32 + hi8];
            v16bf pf = mkfrag(*(const v8bf*)pp, *(const v8bf*)(pp + 16));
#pragma unroll
            for (int d = 0; d < 8; ++d) {
                const bf16_t* vp = &Vs[(d * 16 + lm) * LP + c * 32 + hi8];
                v16bf vf = mkfrag(*(const v8bf*)vp, *(const v8bf*)(vp + 16));
                o[d].v = wmma_bf16(pf, vf, o[d].v);
            }
        }
    }

    // ---- epilogue: normalize and store bf16 rows of [B*T][D] ----
#pragma unroll
    for (int r = 0; r < 8; ++r) {
        float inv = 1.0f / lrow[r];
        int q = iq * 128 + wave * 16 + r + hi8;     // token index in [0,T)
        size_t rowbase = ((size_t)(b * CT + q)) * CD + h * CDK;
#pragma unroll
        for (int d = 0; d < 8; ++d) {
            int dk = d * 16 + lm;
            Out[rowbase + dk] = to_bf16(o[d].f[r] * inv);
        }
    }
}

// ---------------------------------------------------------------------------
// Host-side orchestration
// ---------------------------------------------------------------------------
extern "C" void kernel_launch(void* const* d_in, const int* in_sizes, int n_in,
                              void* d_out, int out_size, void* d_ws, size_t ws_size,
                              hipStream_t stream) {
    (void)in_sizes; (void)n_in; (void)out_size; (void)ws_size;
    const float* x  = (const float*)d_in[0];
    const float* Wq = (const float*)d_in[1];
    const float* Wk = (const float*)d_in[2];
    const float* Wv = (const float*)d_in[3];
    const float* Wo = (const float*)d_in[4];
    // d_in[5] = mask (implicit causal, unused)
    float* out = (float*)d_out;

    char* w = (char*)d_ws;
    bf16_t* xb   = (bf16_t*)w;  w += (size_t)CM * CD * 2;
    bf16_t* wqb  = (bf16_t*)w;  w += (size_t)CD * CD * 2;
    bf16_t* wkb  = (bf16_t*)w;  w += (size_t)CD * CD * 2;
    bf16_t* wvb  = (bf16_t*)w;  w += (size_t)CD * CD * 2;
    bf16_t* wob  = (bf16_t*)w;  w += (size_t)CD * CD * 2;
    bf16_t* qbuf = (bf16_t*)w;  w += (size_t)CB * CH * CT * CDK * 2;
    bf16_t* kbuf = (bf16_t*)w;  w += (size_t)CB * CH * CT * CDK * 2;
    bf16_t* vbuf = (bf16_t*)w;  w += (size_t)CB * CH * CDK * CT * 2;
    bf16_t* abuf = (bf16_t*)w;  w += (size_t)CM * CD * 2;

    // 1) fp32 -> bf16
    {
        int n = CM * CD;
        cvt_f32_bf16<<<(n + 1023) / 1024, 256, 0, stream>>>(x, xb, n);
        n = CD * CD;
        cvt_f32_bf16<<<(n + 1023) / 1024, 256, 0, stream>>>(Wq, wqb, n);
        cvt_f32_bf16<<<(n + 1023) / 1024, 256, 0, stream>>>(Wk, wkb, n);
        cvt_f32_bf16<<<(n + 1023) / 1024, 256, 0, stream>>>(Wv, wvb, n);
        cvt_f32_bf16<<<(n + 1023) / 1024, 256, 0, stream>>>(Wo, wob, n);
    }

    dim3 ggrid(CD / 128, CM / 128);   // (16, 32)
    dim3 gblk(256);

    // 2) projections (RoPE fused for Q/K, transpose for V)
    gemm_bf16<1><<<ggrid, gblk, 0, stream>>>(xb, wqb, (void*)qbuf, CM, CD, CD);
    gemm_bf16<1><<<ggrid, gblk, 0, stream>>>(xb, wkb, (void*)kbuf, CM, CD, CD);
    gemm_bf16<2><<<ggrid, gblk, 0, stream>>>(xb, wvb, (void*)vbuf, CM, CD, CD);

    // 3) flash attention: 512 workgroups, 104448 B dynamic LDS
    {
        constexpr size_t SMEM = (size_t)(128 * 136 + 128 * 136 + 8 * 16 * 136) * 2;
        dim3 agrid(CB * CH * (CT / 128));   // 512
        attn_flash<<<agrid, gblk, SMEM, stream>>>(qbuf, kbuf, vbuf, abuf);
    }

    // 4) output projection -> fp32
    gemm_bf16<0><<<ggrid, gblk, 0, stream>>>(abuf, wob, (void*)out, CM, CD, CD);
}
